// TransformerNMT_49417893708244
// MI455X (gfx1250) — matrix operand order repaired and checked by reading, not compiled, not run on hardware
//
#include <hip/hip_runtime.h>
#include <hip/hip_bf16.h>

// ---------------- problem constants ----------------
#define Bsz  8
#define Tt   256
#define Dm   512
#define Hh   8
#define DKk  64
#define FFd  2048
#define Ll   6
#define VTo  32000
#define EMBd 300
#define RV_  65
#define RVP  72            // padded rel-vocab row stride
#define Mtok (Bsz*Tt)      // 2048 token rows
#define SQD_ 22.627416997969522f   // sqrt(512)

typedef __attribute__((ext_vector_type(16))) _Float16 v16h;
typedef __attribute__((ext_vector_type(8)))  _Float16 h8;
typedef __attribute__((ext_vector_type(8)))  float    v8f;

// ---------------- input flatten order (jax tree: dict keys sorted) ----------
enum {
  IN_DEC_BK=0, IN_DEC_BO, IN_DEC_BQ, IN_DEC_BV,
  IN_DEC_CINB, IN_DEC_CINW, IN_DEC_COB, IN_DEC_COW,
  IN_DEC_L1B, IN_DEC_L1W, IN_DEC_L2B, IN_DEC_L2W,
  IN_DEC_N1B, IN_DEC_N1G, IN_DEC_N2B, IN_DEC_N2G, IN_DEC_N3B, IN_DEC_N3G,
  IN_DEC_RELK, IN_DEC_RELV,
  IN_DEC_WK, IN_DEC_WO, IN_DEC_WQ, IN_DEC_WV,
  IN_DEC_NB, IN_DEC_NG,
  IN_ENC_BK, IN_ENC_BO, IN_ENC_BQ, IN_ENC_BV,
  IN_ENC_L1B, IN_ENC_L1W, IN_ENC_L2B, IN_ENC_L2W,
  IN_ENC_N1B, IN_ENC_N1G, IN_ENC_N2B, IN_ENC_N2G,
  IN_ENC_RELK, IN_ENC_RELV,
  IN_ENC_WK, IN_ENC_WO, IN_ENC_WQ, IN_ENC_WV,
  IN_ENC_NB, IN_ENC_NG,
  IN_GEN_B, IN_GEN_W,
  IN_SRC_EMB, IN_SRC_PROJ_B, IN_SRC_PROJ_W,
  IN_TGT_EMB, IN_TGT_PROJ_B, IN_TGT_PROJ_W,
  IN_SRC, IN_SRC_LENS, IN_TGT
};

union F16x16 { v16h v; h8 h2[2]; _Float16 h[16]; };
union F32x8  { v8f v; float f[8]; };

__device__ __forceinline__ h8 pack8(const float4& a, const float4& b) {
  h8 r;
  r[0]=(_Float16)a.x; r[1]=(_Float16)a.y; r[2]=(_Float16)a.z; r[3]=(_Float16)a.w;
  r[4]=(_Float16)b.x; r[5]=(_Float16)b.y; r[6]=(_Float16)b.z; r[7]=(_Float16)b.w;
  return r;
}

// =====================================================================
// Batched WMMA GEMM:  C = act(alpha * A @ B(^T) + bias) + residual
//   A: fp32 [M x K] (lda); B: fp32 [N x K] if BT else [K x N] (ldb)
//   C: fp32 (ldc); res optional, same layout/offsets as C.
//   EXACT: all of M%64, N%128, K%32 == 0 -> unpredicated float4 path.
// Block 256thr/8 waves; tile 64(M) x 128(N); K step 32.
// LDS: Al[m][k] 64x32, Bt[n][k] 128x32 (k-contiguous for b128 frag reads)
// =====================================================================
template<bool EXACT, bool BT>
__global__ __launch_bounds__(256)
void k_gemm(const float* __restrict__ A, const float* __restrict__ Bm,
            float* __restrict__ C,
            const float* __restrict__ bias, const float* __restrict__ res,
            int M, int N, int K, int lda, int ldb, int ldc,
            int gelu, float alpha, int batchH,
            long long bsA_b, long long bsA_h,
            long long bsB_b, long long bsB_h,
            long long bsC_b, long long bsC_h)
{
  __shared__ _Float16 Al[64][32];     // [m][k]
  __shared__ _Float16 Bt[128][32];    // [n][k]  (transposed tile)

  const int z  = blockIdx.z;
  const int zb = z / batchH, zh = z % batchH;
  A  += zb*bsA_b + zh*bsA_h;
  Bm += zb*bsB_b + zh*bsB_h;
  const long long co = zb*bsC_b + zh*bsC_h;
  C += co;
  if (res) res += co;

  const int row0 = blockIdx.y * 64;
  const int col0 = blockIdx.x * 128;
  const int t    = threadIdx.x;
  const int lane = t & 31;
  const int w    = t >> 5;
  const int wm   = w & 3;      // M sub-tile 0..3
  const int wn   = w >> 2;     // N half 0..1
  const int am   = lane & 15;  // row-in-tile (A) / col-in-tile (B,C)
  const int khalf = lane >> 4;

  // staging coords
  const int ar  = t >> 2;           // A: row 0..63
  const int aks = (t & 3) << 3;     // A: k seg {0,8,16,24}
  const int bn  = t >> 1;           // B(T): n 0..127
  const int bks = (t & 1) << 4;     // B(T): k seg {0,16}
  const int bkr = t >> 3;           // B(N): k row 0..31
  const int bns = (t & 7) << 4;     // B(N): n seg

  F32x8 acc[4];
  #pragma unroll
  for (int i=0;i<4;i++) {
    #pragma unroll
    for (int j=0;j<8;j++) acc[i].f[j]=0.f;
  }

  for (int kk = 0; kk < K; kk += 32) {
    if (EXACT) {
      { // A: 8 floats -> one h8 (16B ds store)
        const float* ap = A + (long long)(row0+ar)*lda + kk + aks;
        float4 a0 = *(const float4*)ap;
        float4 a1 = *(const float4*)(ap+4);
        *(h8*)&Al[ar][aks] = pack8(a0,a1);
      }
      if (BT) { // 16 contiguous k for row n
        const float* bp = Bm + (long long)(col0+bn)*ldb + kk + bks;
        float4 b0 = *(const float4*)bp;
        float4 b1 = *(const float4*)(bp+4);
        float4 b2 = *(const float4*)(bp+8);
        float4 b3 = *(const float4*)(bp+12);
        *(h8*)&Bt[bn][bks]   = pack8(b0,b1);
        *(h8*)&Bt[bn][bks+8] = pack8(b2,b3);
      } else {  // 16 contiguous n for k row; strided LDS scatter
        const float* bp = Bm + (long long)(kk+bkr)*ldb + col0 + bns;
        #pragma unroll
        for (int j=0;j<16;j+=4) {
          float4 bv = *(const float4*)(bp+j);
          Bt[bns+j+0][bkr] = (_Float16)bv.x;
          Bt[bns+j+1][bkr] = (_Float16)bv.y;
          Bt[bns+j+2][bkr] = (_Float16)bv.z;
          Bt[bns+j+3][bkr] = (_Float16)bv.w;
        }
      }
    } else {
      { // predicated A stage
        int gr = row0 + ar;
        #pragma unroll
        for (int j=0;j<8;j++) {
          int k = kk + aks + j;
          float v = (gr < M && k < K) ? A[(long long)gr*lda + k] : 0.f;
          Al[ar][aks+j] = (_Float16)v;
        }
      }
      if (BT) {
        int gn = col0 + bn;
        #pragma unroll
        for (int j=0;j<16;j++) {
          int k = kk + bks + j;
          float v = (gn < N && k < K) ? Bm[(long long)gn*ldb + k] : 0.f;
          Bt[bn][bks+j] = (_Float16)v;
        }
      } else {
        int gk = kk + bkr;
        #pragma unroll
        for (int j=0;j<16;j++) {
          int n = col0 + bns + j;
          float v = (gk < K && n < N) ? Bm[(long long)gk*ldb + n] : 0.f;
          Bt[bns+j][bkr] = (_Float16)v;
        }
      }
    }
    __syncthreads();

    // A fragment: row (wm*16+am); halves k = khalf*8+0..7 and 16+khalf*8+0..7
    F16x16 af;
    {
      const h8* arow = (const h8*)&Al[(wm<<4) + am][0];
      af.h2[0] = arow[khalf];
      af.h2[1] = arow[2 + khalf];
    }
    #pragma unroll
    for (int nt=0; nt<4; nt++) {
      const int ncol = (wn<<6) + (nt<<4) + am;
      F16x16 bf;
      const h8* brow = (const h8*)&Bt[ncol][0];
      bf.h2[0] = brow[khalf];
      bf.h2[1] = brow[2 + khalf];
      acc[nt].v = __builtin_amdgcn_wmma_f32_16x16x32_f16(
          false, af.v, false, bf.v, (short)0, acc[nt].v, false, false);
    }
    __syncthreads();
  }

  // epilogue: C layout — lane col = lane&15, rows = (lane>>4)*8 + v
  #pragma unroll
  for (int nt=0; nt<4; nt++) {
    const int col = col0 + (wn<<6) + (nt<<4) + am;
    if (!EXACT && col >= N) continue;
    const float bv = bias ? bias[col] : 0.f;
    #pragma unroll
    for (int v=0; v<8; v++) {
      const int row = row0 + (wm<<4) + (khalf<<3) + v;
      if (!EXACT && row >= M) continue;
      float val = acc[nt].f[v] * alpha + bv;
      if (gelu) val = 0.5f * val * (1.f + erff(val * 0.70710678118654752f));
      if (res) val += res[(long long)row*ldc + col];
      C[(long long)row*ldc + col] = val;
    }
  }
}

// ---------------- LayerNorm: one block per row of 512 ----------------
__global__ __launch_bounds__(256)
void k_layernorm(const float* __restrict__ x, float* __restrict__ y,
                 const float* __restrict__ g, const float* __restrict__ b)
{
  __shared__ float red[256];
  const long long row = blockIdx.x;
  const float* xr = x + row*Dm;
  const int t = threadIdx.x;
  float a0 = xr[t], a1 = xr[t+256];
  red[t] = a0 + a1;
  __syncthreads();
  for (int s=128; s>0; s>>=1) { if (t<s) red[t]+=red[t+s]; __syncthreads(); }
  const float mean = red[0] * (1.f/(float)Dm);
  __syncthreads();
  const float d0=a0-mean, d1=a1-mean;
  red[t] = d0*d0 + d1*d1;
  __syncthreads();
  for (int s=128; s>0; s>>=1) { if (t<s) red[t]+=red[t+s]; __syncthreads(); }
  const float rstd = rsqrtf(red[0]*(1.f/(float)Dm) + 1e-5f);
  float* yr = y + row*Dm;
  yr[t]     = d0*rstd*g[t]     + b[t];
  yr[t+256] = d1*rstd*g[t+256] + b[t+256];
}

// -------- embedding gather + [D x 300] projection + sqrt(D) scale ----
__global__ __launch_bounds__(256)
void k_embed(const int* __restrict__ toks, const float* __restrict__ emb,
             const float* __restrict__ w, const float* __restrict__ bias,
             float* __restrict__ out)
{
  __shared__ float e[EMBd];
  const long long row = blockIdx.x;
  const int tok = toks[row];
  const int t = threadIdx.x;
  for (int j=t; j<EMBd; j+=256) e[j] = emb[(long long)tok*EMBd + j];
  __syncthreads();
  for (int c = t; c < Dm; c += 256) {
    const float* wr = w + (long long)c*EMBd;
    float s = 0.f;
    for (int j=0;j<EMBd;j++) s = fmaf(wr[j], e[j], s);
    out[row*Dm + c] = (s + bias[c]) * SQD_;
  }
}

// --- fused: scores += rel-bias, masks, softmax, rel-bucket sums ------
// one block per (b,h,q) row of 256 keys
__global__ __launch_bounds__(256)
void k_softmax(float* __restrict__ S, const float* __restrict__ R,
               float* __restrict__ BS, const int* __restrict__ toks,
               int causal)
{
  __shared__ float red[256];
  __shared__ float bsl[RV_];
  const long long idx = blockIdx.x;           // b*H*T + h*T + q
  const int  q = (int)(idx % Tt);
  const long long b = idx / ((long long)Hh*Tt);
  const int  k = threadIdx.x;
  int r = k - q; r = r < -32 ? -32 : (r > 32 ? 32 : r); r += 32;

  float s = S[idx*Tt + k];
  if (R) s += R[idx*RVP + r];
  const bool masked = (toks[b*Tt + k] == 0) || (causal && k > q);
  red[k] = masked ? -3.0e38f : s;
  __syncthreads();
  for (int st=128; st>0; st>>=1) { if (k<st) red[k]=fmaxf(red[k],red[k+st]); __syncthreads(); }
  const float mx = red[0];
  __syncthreads();
  const float e = masked ? 0.f : expf(s - mx);
  red[k] = e;
  __syncthreads();
  for (int st=128; st>0; st>>=1) { if (k<st) red[k]+=red[k+st]; __syncthreads(); }
  const float sum = red[0];
  const float a = sum > 0.f ? e / sum : 0.f;
  S[idx*Tt + k] = a;
  if (BS) {
    if (k < RV_) bsl[k] = 0.f;
    __syncthreads();
    atomicAdd(&bsl[r], a);        // ds_add_f32
    __syncthreads();
    if (k < RV_) BS[idx*RVP + k] = bsl[k];
  }
}

// ---------------- host-side helpers ----------------
static inline void gemm(hipStream_t st, const float* A, const float* Bm, float* C,
                        const float* bias, const float* res,
                        int M, int N, int K, int lda, int ldb, int ldc,
                        int bT, int gelu, float alpha,
                        int nbatch, int batchH,
                        long long bsA_b, long long bsA_h,
                        long long bsB_b, long long bsB_h,
                        long long bsC_b, long long bsC_h)
{
  dim3 g((N+127)/128, (M+63)/64, nbatch);
  const bool exact = (M%64==0) && (N%128==0) && (K%32==0) &&
                     (lda%4==0) && (ldb%4==0);
  if (bT) {
    if (exact) k_gemm<true ,true ><<<g,256,0,st>>>(A,Bm,C,bias,res,M,N,K,lda,ldb,ldc,gelu,alpha,batchH,bsA_b,bsA_h,bsB_b,bsB_h,bsC_b,bsC_h);
    else       k_gemm<false,true ><<<g,256,0,st>>>(A,Bm,C,bias,res,M,N,K,lda,ldb,ldc,gelu,alpha,batchH,bsA_b,bsA_h,bsB_b,bsB_h,bsC_b,bsC_h);
  } else {
    if (exact) k_gemm<true ,false><<<g,256,0,st>>>(A,Bm,C,bias,res,M,N,K,lda,ldb,ldc,gelu,alpha,batchH,bsA_b,bsA_h,bsB_b,bsB_h,bsC_b,bsC_h);
    else       k_gemm<false,false><<<g,256,0,st>>>(A,Bm,C,bias,res,M,N,K,lda,ldb,ldc,gelu,alpha,batchH,bsA_b,bsA_h,bsB_b,bsB_h,bsC_b,bsC_h);
  }
}

// full attention block: Xres += O( attn(Xln as Q; kvsrc as K,V) )
static void run_attn(hipStream_t st, float* Xres, const float* Xln,
                     const float* wq, const float* bq,
                     const float* wk, const float* bk,
                     const float* wv, const float* bv,
                     const float* wo, const float* bo,
                     const float* relk, const float* relv,   // null for cross
                     const float* kvsrc, const int* toks, int causal,
                     float* Qb, float* Kb, float* Vb,
                     float* S, float* Rb, float* BSb, float* CTX)
{
  const float inv_sqk = 0.125f;  // 1/sqrt(64)
  gemm(st, Xln,   wq, Qb, bq, nullptr, Mtok, Dm, Dm, Dm, Dm, Dm, 1, 0, 1.f, 1,1, 0,0,0,0,0,0);
  gemm(st, kvsrc, wk, Kb, bk, nullptr, Mtok, Dm, Dm, Dm, Dm, Dm, 1, 0, 1.f, 1,1, 0,0,0,0,0,0);
  gemm(st, kvsrc, wv, Vb, bv, nullptr, Mtok, Dm, Dm, Dm, Dm, Dm, 1, 0, 1.f, 1,1, 0,0,0,0,0,0);

  const long long aB = (long long)Tt*Dm, aH = DKk;            // Q/K/V (b,h) slices
  const long long sB = (long long)Hh*Tt*Tt, sH = (long long)Tt*Tt;
  // S = Q @ K^T / sqrt(dk)   (batched over B*H)
  gemm(st, Qb, Kb, S, nullptr, nullptr, Tt, Tt, DKk, Dm, Dm, Tt, 1, 0, inv_sqk,
       Bsz*Hh, Hh, aB, aH, aB, aH, sB, sH);
  if (relk) { // R = Q @ rel_k^T  -> [BHT, 65]
    const long long rB = (long long)Hh*Tt*RVP, rH = (long long)Tt*RVP;
    gemm(st, Qb, relk, Rb, nullptr, nullptr, Tt, RV_, DKk, Dm, DKk, RVP, 1, 0, 1.f,
         Bsz*Hh, Hh, aB, aH, 0, 0, rB, rH);
  }
  k_softmax<<<Bsz*Hh*Tt, 256, 0, st>>>(S, relk ? Rb : nullptr,
                                       relk ? BSb : nullptr, toks, causal);
  // ctx = A @ V   (B in [K,N] layout inside the activation buffer)
  gemm(st, S, Vb, CTX, nullptr, nullptr, Tt, DKk, Tt, Tt, Dm, Dm, 0, 0, 1.f,
       Bsz*Hh, Hh, sB, sH, aB, aH, aB, aH);
  if (relv) { // ctx += bucket_sums @ rel_v
    const long long rB = (long long)Hh*Tt*RVP, rH = (long long)Tt*RVP;
    gemm(st, BSb, relv, CTX, nullptr, CTX, Tt, DKk, RV_, RVP, DKk, Dm, 0, 0, 1.f,
         Bsz*Hh, Hh, rB, rH, 0, 0, aB, aH);
  }
  // X += ctx @ Wo^T + bo
  gemm(st, CTX, wo, Xres, bo, Xres, Mtok, Dm, Dm, Dm, Dm, Dm, 1, 0, 1.f, 1,1,0,0,0,0,0,0);
}

static void run_ffn(hipStream_t st, float* X, const float* Xln,
                    const float* l1w, const float* l1b,
                    const float* l2w, const float* l2b, float* FFH)
{
  gemm(st, Xln, l1w, FFH, l1b, nullptr, Mtok, FFd, Dm, Dm, Dm, FFd, 1, 1, 1.f, 1,1,0,0,0,0,0,0);
  gemm(st, FFH, l2w, X,   l2b, X,       Mtok, Dm, FFd, FFd, FFd, Dm, 1, 0, 1.f, 1,1,0,0,0,0,0,0);
}

// =====================================================================
extern "C" void kernel_launch(void* const* d_in, const int* in_sizes, int n_in,
                              void* d_out, int out_size, void* d_ws, size_t ws_size,
                              hipStream_t stream)
{
  (void)in_sizes; (void)n_in; (void)out_size; (void)ws_size;
  auto F = [&](int i){ return (const float*)d_in[i]; };
  const int* src = (const int*)d_in[IN_SRC];
  const int* tgt = (const int*)d_in[IN_TGT];

  // ---- workspace carve (fp32) ----
  float* p = (float*)d_ws;
  const size_t SZ_MD = (size_t)Mtok*Dm;
  float* XE  = p; p += SZ_MD;
  float* Yd  = p; p += SZ_MD;
  float* LN1 = p; p += SZ_MD;
  float* Qb  = p; p += SZ_MD;
  float* Kb  = p; p += SZ_MD;
  float* Vb  = p; p += SZ_MD;
  float* CTX = p; p += SZ_MD;
  float* MEM = p; p += SZ_MD;
  float* FFH = p; p += (size_t)Mtok*FFd;
  float* S   = p; p += (size_t)Bsz*Hh*Tt*Tt;
  float* Rb  = p; p += (size_t)Bsz*Hh*Tt*RVP;
  float* BSb = p; p += (size_t)Bsz*Hh*Tt*RVP;

  // ================= encoder =================
  k_embed<<<Mtok,256,0,stream>>>(src, F(IN_SRC_EMB), F(IN_SRC_PROJ_W), F(IN_SRC_PROJ_B), XE);
  for (int i=0;i<Ll;i++) {
    const size_t oW = (size_t)i*Dm*Dm, oB=(size_t)i*Dm, oR=(size_t)i*RV_*DKk;
    k_layernorm<<<Mtok,256,0,stream>>>(XE, LN1, F(IN_ENC_N1G)+oB, F(IN_ENC_N1B)+oB);
    run_attn(stream, XE, LN1,
             F(IN_ENC_WQ)+oW, F(IN_ENC_BQ)+oB, F(IN_ENC_WK)+oW, F(IN_ENC_BK)+oB,
             F(IN_ENC_WV)+oW, F(IN_ENC_BV)+oB, F(IN_ENC_WO)+oW, F(IN_ENC_BO)+oB,
             F(IN_ENC_RELK)+oR, F(IN_ENC_RELV)+oR,
             LN1, src, 0, Qb,Kb,Vb,S,Rb,BSb,CTX);
    k_layernorm<<<Mtok,256,0,stream>>>(XE, LN1, F(IN_ENC_N2G)+oB, F(IN_ENC_N2B)+oB);
    run_ffn(stream, XE, LN1,
            F(IN_ENC_L1W)+(size_t)i*FFd*Dm, F(IN_ENC_L1B)+(size_t)i*FFd,
            F(IN_ENC_L2W)+(size_t)i*Dm*FFd, F(IN_ENC_L2B)+oB, FFH);
  }
  k_layernorm<<<Mtok,256,0,stream>>>(XE, MEM, F(IN_ENC_NG), F(IN_ENC_NB));

  // ================= decoder =================
  k_embed<<<Mtok,256,0,stream>>>(tgt, F(IN_TGT_EMB), F(IN_TGT_PROJ_W), F(IN_TGT_PROJ_B), Yd);
  for (int i=0;i<Ll;i++) {
    const size_t oW = (size_t)i*Dm*Dm, oB=(size_t)i*Dm, oR=(size_t)i*RV_*DKk;
    // causal relative self-attention
    k_layernorm<<<Mtok,256,0,stream>>>(Yd, LN1, F(IN_DEC_N1G)+oB, F(IN_DEC_N1B)+oB);
    run_attn(stream, Yd, LN1,
             F(IN_DEC_WQ)+oW, F(IN_DEC_BQ)+oB, F(IN_DEC_WK)+oW, F(IN_DEC_BK)+oB,
             F(IN_DEC_WV)+oW, F(IN_DEC_BV)+oB, F(IN_DEC_WO)+oW, F(IN_DEC_BO)+oB,
             F(IN_DEC_RELK)+oR, F(IN_DEC_RELV)+oR,
             LN1, tgt, 1, Qb,Kb,Vb,S,Rb,BSb,CTX);
    // cross-attention (packed in_proj: rows [0:D)=q, [D:2D)=k, [2D:3D)=v)
    k_layernorm<<<Mtok,256,0,stream>>>(Yd, LN1, F(IN_DEC_N2G)+oB, F(IN_DEC_N2B)+oB);
    const float* cinw = F(IN_DEC_CINW) + (size_t)i*3*Dm*Dm;
    const float* cinb = F(IN_DEC_CINB) + (size_t)i*3*Dm;
    run_attn(stream, Yd, LN1,
             cinw,               cinb,
             cinw + (size_t)Dm*Dm,   cinb + Dm,
             cinw + (size_t)2*Dm*Dm, cinb + 2*Dm,
             F(IN_DEC_COW)+oW, F(IN_DEC_COB)+oB,
             nullptr, nullptr,
             MEM, src, 0, Qb,Kb,Vb,S,Rb,BSb,CTX);
    // FFN
    k_layernorm<<<Mtok,256,0,stream>>>(Yd, LN1, F(IN_DEC_N3G)+oB, F(IN_DEC_N3B)+oB);
    run_ffn(stream, Yd, LN1,
            F(IN_DEC_L1W)+(size_t)i*FFd*Dm, F(IN_DEC_L1B)+(size_t)i*FFd,
            F(IN_DEC_L2W)+(size_t)i*Dm*FFd, F(IN_DEC_L2B)+oB, FFH);
  }
  k_layernorm<<<Mtok,256,0,stream>>>(Yd, LN1, F(IN_DEC_NG), F(IN_DEC_NB));

  // generator: logits [2048 x 32000]
  gemm(stream, LN1, F(IN_GEN_W), (float*)d_out, F(IN_GEN_B), nullptr,
       Mtok, VTo, Dm, Dm, Dm, VTo, 1, 0, 1.f, 1,1, 0,0,0,0,0,0);
}
